// ParallelMLP_11793980195162
// MI455X (gfx1250) — compile-verified, matrix operand order checked by reading, not run on hardware
//
#include <hip/hip_runtime.h>

// Problem constants (from the reference)
#define NTOK 8192
#define DMODEL 2048
#define HDIM 2048
#define NEXP 8
#define TOPK 2
#define NROWS (NTOK * TOPK)   // 16384 permuted rows

typedef __bf16 bf16x16 __attribute__((ext_vector_type(16)));
typedef float  f32x8   __attribute__((ext_vector_type(8)));

struct Frag { union { uint4 u[2]; bf16x16 v; }; };

// fp32 -> bf16 with round-half-up bias (cheap: 1 VALU op), pair-pack via v_perm_b32
static __device__ __forceinline__ unsigned short f2bf(float f) {
    unsigned u = __builtin_bit_cast(unsigned, f) + 0x8000u;
    return (unsigned short)(u >> 16);
}
static __device__ __forceinline__ unsigned pack2(float lo, float hi) {
    unsigned a = __builtin_bit_cast(unsigned, lo) + 0x8000u;
    unsigned b = __builtin_bit_cast(unsigned, hi) + 0x8000u;
    // result bytes: [0,1]=a[2,3] (lo bf16), [2,3]=b[2,3] (hi bf16)
    return __builtin_amdgcn_perm(b, a, 0x07060302u);
}
static __device__ __forceinline__ f32x8 wmma_bf16(bf16x16 a, bf16x16 b, f32x8 c) {
    return __builtin_amdgcn_wmma_f32_16x16x32_bf16(false, a, false, b, (short)0, c, false, false);
}
static __device__ __forceinline__ void ldfrag(Frag& f, const unsigned short* p) {
    f.u[0] = *(const uint4*)(p);        // ds_load_b128: K = 8*kh .. +7
    f.u[1] = *(const uint4*)(p + 16);   // ds_load_b128: K = 16+8*kh .. +7
}

// async global->LDS copy (CDNA5): IOFFSET applies to BOTH lds and global addr
#define A_ASYNC_LD128(lds_u32, gptr, OFF)                                        \
    asm volatile("global_load_async_to_lds_b128 %0, %1, off offset:" OFF         \
                 :: "v"(lds_u32), "v"(gptr) : "memory")
#define WAIT_ASYNC0() asm volatile("s_wait_asynccnt 0x0" ::: "memory")

// ---------------- routing ----------------
__global__ void route_offsets_k(const int* __restrict__ bspe,
                                int* __restrict__ offsets, int* __restrict__ counters) {
    if (threadIdx.x == 0) {
        int acc = 0;
        for (int e = 0; e < NEXP; ++e) { offsets[e] = acc; acc += bspe[e]; }
        offsets[NEXP] = acc;
    }
    if (threadIdx.x < NEXP) counters[threadIdx.x] = 0;
}

__global__ void route_scatter_k(const int* __restrict__ eidx, const float* __restrict__ ew,
                                const int* __restrict__ offsets, int* __restrict__ counters,
                                int* __restrict__ rowToken, float* __restrict__ rowWeight) {
    int i = blockIdx.x * blockDim.x + threadIdx.x;
    if (i < NROWS) {
        int e = eidx[i];
        int slot = atomicAdd(&counters[e], 1);
        int r = offsets[e] + slot;
        rowToken[r]  = i / TOPK;
        rowWeight[r] = ew[i];
    }
}

// ---------------- stage 1: H = silu(Xg@w1) * (Xg@w3), per expert ----------------
// block tile 128(M) x 64(N), BK=32; 4 waves (2x2), wave tile 64x32; double-buffered LDS
__global__ __launch_bounds__(128)
void moe_stage1_k(const float* __restrict__ x, const float* __restrict__ w1,
                  const float* __restrict__ w3, const int* __restrict__ offsets,
                  const int* __restrict__ rowToken, unsigned short* __restrict__ Hmat) {
    const int e = blockIdx.z;
    const int rowStart = offsets[e];
    const int Me = offsets[e + 1] - rowStart;
    const int m0 = blockIdx.y * 128;
    if (m0 >= Me) return;
    const int n0 = blockIdx.x * 64;

    __shared__ unsigned short As[2 * 128 * 32];
    __shared__ unsigned short B1s[2 * 64 * 32];
    __shared__ unsigned short B3s[2 * 64 * 32];

    const int tid  = threadIdx.x;
    const int wid  = tid >> 5, lane = tid & 31;
    const int lm   = lane & 15, kh = lane >> 4;
    const int wrow = (wid >> 1) * 64, wcol = (wid & 1) * 32;

    // clamp to valid addresses; garbage values only reach guarded (discarded) rows
    const int myrow = m0 + tid;
    int tok = rowToken[rowStart + (myrow < Me ? myrow : 0)];
    tok = (tok < 0) ? 0 : (tok >= NTOK ? NTOK - 1 : tok);
    const float* xrow = x + (size_t)tok * DMODEL;

    const float* w1e = w1 + (size_t)e * DMODEL * HDIM;
    const float* w3e = w3 + (size_t)e * DMODEL * HDIM;
    const int kq = tid >> 4, nq = tid & 15;     // B staging: 4 k-rows x 4 n-cols

    f32x8 acc1[4][2], acc3[4][2];
#pragma unroll
    for (int a = 0; a < 4; ++a)
#pragma unroll
        for (int b = 0; b < 2; ++b)
#pragma unroll
            for (int j = 0; j < 8; ++j) { acc1[a][b][j] = 0.f; acc3[a][b][j] = 0.f; }

    uint4 av[4];                 // packed bf16 A row (32 values)
    unsigned pb1[8], pb3[8];     // packed bf16 B words

    auto loadA = [&](int kt) {
        const float4* p = (const float4*)(xrow + kt * 32);
#pragma unroll
        for (int i = 0; i < 4; ++i) {
            float4 a = p[2 * i], b = p[2 * i + 1];
            av[i].x = pack2(a.x, a.y); av[i].y = pack2(a.z, a.w);
            av[i].z = pack2(b.x, b.y); av[i].w = pack2(b.z, b.w);
        }
    };
    auto loadB = [&](int kt) {
        float4 f1[4], f3[4];
#pragma unroll
        for (int dk = 0; dk < 4; ++dk) {
            size_t off = (size_t)(kt * 32 + kq * 4 + dk) * HDIM + (n0 + 4 * nq);
            f1[dk] = *(const float4*)(w1e + off);
            f3[dk] = *(const float4*)(w3e + off);
        }
#pragma unroll
        for (int dn = 0; dn < 4; ++dn)
#pragma unroll
            for (int p = 0; p < 2; ++p) {
                pb1[dn * 2 + p] = pack2(((const float*)&f1[2 * p])[dn], ((const float*)&f1[2 * p + 1])[dn]);
                pb3[dn * 2 + p] = pack2(((const float*)&f3[2 * p])[dn], ((const float*)&f3[2 * p + 1])[dn]);
            }
    };
    auto storeTiles = [&](int buf) {
        uint4* asr = (uint4*)(As + buf * (128 * 32) + tid * 32);
#pragma unroll
        for (int i = 0; i < 4; ++i) asr[i] = av[i];
        unsigned* b1u = (unsigned*)(B1s + buf * (64 * 32));
        unsigned* b3u = (unsigned*)(B3s + buf * (64 * 32));
#pragma unroll
        for (int dn = 0; dn < 4; ++dn) {
            int idx = (4 * nq + dn) * 16 + 2 * kq;
#pragma unroll
            for (int p = 0; p < 2; ++p) {
                b1u[idx + p] = pb1[dn * 2 + p];
                b3u[idx + p] = pb3[dn * 2 + p];
            }
        }
    };

    loadA(0); loadB(0); storeTiles(0);

    const int KT = DMODEL / 32;
    for (int kt = 0; kt < KT; ++kt) {
        const int cur = kt & 1, nxt = cur ^ 1;
        __syncthreads();
        const bool more = (kt + 1 < KT);
        if (more) {
            __builtin_prefetch((const void*)(xrow + (kt + 1) * 32), 0, 3);
            loadA(kt + 1);
            loadB(kt + 1);
        }
        // ---- compute on buffer `cur`
        const unsigned short* Ab  = As  + cur * (128 * 32);
        const unsigned short* B1b = B1s + cur * (64 * 32);
        const unsigned short* B3b = B3s + cur * (64 * 32);
        Frag b1[2], b3[2];
#pragma unroll
        for (int nf = 0; nf < 2; ++nf) {
            int nb = wcol + nf * 16 + lm;
            ldfrag(b1[nf], B1b + nb * 32 + 8 * kh);
            ldfrag(b3[nf], B3b + nb * 32 + 8 * kh);
        }
#pragma unroll
        for (int mf = 0; mf < 4; ++mf) {
            Frag a;
            ldfrag(a, Ab + (wrow + mf * 16 + lm) * 32 + 8 * kh);
#pragma unroll
            for (int nf = 0; nf < 2; ++nf) {
                acc1[mf][nf] = wmma_bf16(a.v, b1[nf].v, acc1[mf][nf]);
                acc3[mf][nf] = wmma_bf16(a.v, b3[nf].v, acc3[mf][nf]);
            }
        }
        if (more) storeTiles(nxt);
    }

    // ---- fused SwiGLU epilogue -> bf16 Hmat
#pragma unroll
    for (int mf = 0; mf < 4; ++mf) {
#pragma unroll
        for (int j = 0; j < 8; ++j) {
            int row = m0 + wrow + mf * 16 + j + 8 * kh;
            if (row < Me) {
                size_t base = (size_t)(rowStart + row) * HDIM + n0 + wcol + lm;
#pragma unroll
                for (int nf = 0; nf < 2; ++nf) {
                    float a1 = acc1[mf][nf][j], a3 = acc3[mf][nf][j];
                    float hv = (a1 / (1.f + __expf(-a1))) * a3;   // silu(a1)*a3
                    Hmat[base + nf * 16] = f2bf(hv);
                }
            }
        }
    }
}

// ---------------- stage 2: out[tok] += w * (Hg @ w2), per expert ----------------
// block tile 128x128, BK=32; 4 waves (2x2), wave tile 64x64; double-buffered LDS,
// A tile streamed with CDNA5 async global->LDS (ASYNCcnt), B tile via VGPR convert.
__global__ __launch_bounds__(128)
void moe_stage2_k(const unsigned short* __restrict__ Hmat, const float* __restrict__ w2,
                  const int* __restrict__ offsets, const int* __restrict__ rowToken,
                  const float* __restrict__ rowWeight, float* __restrict__ out) {
    const int e = blockIdx.z;
    const int rowStart = offsets[e];
    const int Me = offsets[e + 1] - rowStart;
    const int m0 = blockIdx.y * 128;
    if (m0 >= Me) return;
    const int n0 = blockIdx.x * 128;

    __shared__ unsigned short As[2 * 128 * 32];
    __shared__ unsigned short Bs[2 * 128 * 32];

    const int tid  = threadIdx.x;
    const int wid  = tid >> 5, lane = tid & 31;
    const int lm   = lane & 15, kh = lane >> 4;
    const int wrow = (wid >> 1) * 64, wcol = (wid & 1) * 64;

    const int myrow = m0 + tid;
    int hr = rowStart + myrow; if (hr >= NROWS) hr = NROWS - 1;   // valid addr; values guarded
    const unsigned short* hrow = Hmat + (size_t)hr * HDIM;

    const float* w2e = w2 + (size_t)e * HDIM * DMODEL;
    const int kq = tid >> 5, nq = tid & 31;     // B staging: 8 k-rows x 4 n-cols

    // LDS byte address of this thread's A row (generic-pointer low 32 bits = LDS offset)
    const unsigned ldsA = (unsigned)(uintptr_t)(&As[0]) + (unsigned)(tid * 64);

    f32x8 acc[4][4];
#pragma unroll
    for (int a = 0; a < 4; ++a)
#pragma unroll
        for (int b = 0; b < 4; ++b)
#pragma unroll
            for (int j = 0; j < 8; ++j) acc[a][b][j] = 0.f;

    unsigned pbB[16];

    auto asyncA = [&](int kt, int buf) {       // 64B row: 4x async b128, no VGPR data
        const unsigned short* g = hrow + kt * 32;
        unsigned l = ldsA + (unsigned)(buf * (128 * 64));
        A_ASYNC_LD128(l, g, "0");
        A_ASYNC_LD128(l, g, "16");
        A_ASYNC_LD128(l, g, "32");
        A_ASYNC_LD128(l, g, "48");
    };
    auto loadB = [&](int kt) {
        float4 f[8];
#pragma unroll
        for (int dk = 0; dk < 8; ++dk)
            f[dk] = *(const float4*)(w2e + (size_t)(kt * 32 + kq * 8 + dk) * DMODEL + (n0 + 4 * nq));
#pragma unroll
        for (int dn = 0; dn < 4; ++dn)
#pragma unroll
            for (int p = 0; p < 4; ++p)
                pbB[dn * 4 + p] = pack2(((const float*)&f[2 * p])[dn], ((const float*)&f[2 * p + 1])[dn]);
    };
    auto storeB = [&](int buf) {
        unsigned* bu = (unsigned*)(Bs + buf * (128 * 32));
#pragma unroll
        for (int dn = 0; dn < 4; ++dn) {
            int idx = (4 * nq + dn) * 16 + kq * 4;
#pragma unroll
            for (int p = 0; p < 4; ++p) bu[idx + p] = pbB[dn * 4 + p];
        }
    };

    asyncA(0, 0); loadB(0); storeB(0);

    const int KT = HDIM / 32;
    for (int kt = 0; kt < KT; ++kt) {
        const int cur = kt & 1, nxt = cur ^ 1;
        WAIT_ASYNC0();                 // our async writes into buf `cur` are done
        __syncthreads();               // everyone's tile `cur` is visible
        const bool more = (kt + 1 < KT);
        if (more) {
            asyncA(kt + 1, nxt);       // overlaps with WMMA below
            loadB(kt + 1);
        }
        const unsigned short* Ab = As + cur * (128 * 32);
        const unsigned short* Bb = Bs + cur * (128 * 32);
        Frag bf[4];
#pragma unroll
        for (int nf = 0; nf < 4; ++nf)
            ldfrag(bf[nf], Bb + (wcol + nf * 16 + lm) * 32 + 8 * kh);
#pragma unroll
        for (int mf = 0; mf < 4; ++mf) {
            Frag a;
            ldfrag(a, Ab + (wrow + mf * 16 + lm) * 32 + 8 * kh);
#pragma unroll
            for (int nf = 0; nf < 4; ++nf)
                acc[mf][nf] = wmma_bf16(a.v, bf[nf].v, acc[mf][nf]);
        }
        if (more) storeB(nxt);
    }

    // ---- scatter-combine epilogue: out[token] += weight * val  (each elem hit TOPK times)
#pragma unroll
    for (int mf = 0; mf < 4; ++mf) {
#pragma unroll
        for (int j = 0; j < 8; ++j) {
            int row = m0 + wrow + mf * 16 + j + 8 * kh;
            if (row < Me) {
                int   tok = rowToken[rowStart + row];
                float wgt = rowWeight[rowStart + row];
                float* obase = out + (size_t)tok * DMODEL + n0 + wcol + lm;
#pragma unroll
                for (int nf = 0; nf < 4; ++nf)
                    unsafeAtomicAdd(obase + nf * 16, wgt * acc[mf][nf][j]);
            }
        }
    }
}

// ---------------- launcher ----------------
extern "C" void kernel_launch(void* const* d_in, const int* in_sizes, int n_in,
                              void* d_out, int out_size, void* d_ws, size_t ws_size,
                              hipStream_t stream) {
    const float* x    = (const float*)d_in[0];
    const float* ew   = (const float*)d_in[1];
    const int*   ei   = (const int*)d_in[2];
    const int*   bspe = (const int*)d_in[3];
    const float* w1   = (const float*)d_in[4];
    const float* w2   = (const float*)d_in[5];
    const float* w3   = (const float*)d_in[6];
    float* out = (float*)d_out;

    char* ws = (char*)d_ws;
    int*   offsets   = (int*)ws;                       // E+1 ints
    int*   counters  = (int*)(ws + 64);                // E ints
    int*   rowToken  = (int*)(ws + 256);               // NROWS ints
    float* rowWeight = (float*)(ws + 256 + NROWS * 4); // NROWS floats
    unsigned short* Hmat = (unsigned short*)(ws + (1u << 20)); // NROWS*HDIM bf16 (64 MB)

    hipMemsetAsync(out, 0, sizeof(float) * (size_t)NTOK * DMODEL, stream);

    route_offsets_k<<<1, 64, 0, stream>>>(bspe, offsets, counters);
    route_scatter_k<<<NROWS / 256, 256, 0, stream>>>(ei, ew, offsets, counters, rowToken, rowWeight);

    dim3 g1(HDIM / 64, NROWS / 128, NEXP);     // 32 x 128 x 8 (early-exit over-launch in y)
    moe_stage1_k<<<g1, 128, 0, stream>>>(x, w1, w3, offsets, rowToken, Hmat);

    dim3 g2(DMODEL / 128, NROWS / 128, NEXP);  // 16 x 128 x 8
    moe_stage2_k<<<g2, 128, 0, stream>>>(Hmat, w2, offsets, rowToken, rowWeight, out);
}
